// NER_14809047236793
// MI455X (gfx1250) — matrix operand order repaired and checked by reading, not compiled, not run on hardware
//
#include <hip/hip_runtime.h>
#include <hip/hip_bf16.h>

// ---------------------------------------------------------------------------
// BiGRU + CRF NLL for MI455X (gfx1250, wave32, WMMA + async global->LDS DMA).
// Phase 1: xp = bert_embed @ [W_ih_f;W_ih_b]^T + bias (bf16 WMMA, async-staged,
//          double-buffered K tiles tracked by ASYNCcnt)
// Phase 2: GRU scan fwd+bwd; W_hh in register B-fragments; next step's xp slab
//          async-prefetched into LDS while current step computes
// Phase 3: emissions = h_cat @ W_dense^T
// Phase 4: CRF gold score + forward algorithm, mean NLL -> scalar
// ---------------------------------------------------------------------------

#define Bc 128
#define Sc 256
#define Ec 768
#define Hc 64
#define Tc 9
#define Mc (Bc * Sc)      // 32768 token rows
#define Gc 384            // 2 * 3H concatenated gate dim

typedef __attribute__((ext_vector_type(16))) __bf16 v16bf;
typedef __attribute__((ext_vector_type(8)))  float  v8f;

// Async DMA: copy 16B per lane from global to LDS. VDST takes the LDS byte
// offset == low 32 bits of the generic pointer (ISA 10.2 aperture rule).
__device__ __forceinline__ void async_ld16(void* lds, const void* g)
{
    unsigned l = (unsigned)(uintptr_t)lds;
    asm volatile("global_load_async_to_lds_b128 %0, %1, off"
                 :: "v"(l), "v"(g) : "memory");
}
__device__ __forceinline__ void wait_async0()
{
    asm volatile("s_wait_asynccnt 0" ::: "memory");
}

// Build a v16bf fragment from two runs of 8 fp32 values (LDS or global).
__device__ __forceinline__ v16bf cvt16(const float* p0, const float* p1)
{
    v16bf r;
    #pragma unroll
    for (int i = 0; i < 8; ++i) {
        r[i]     = (__bf16)p0[i];
        r[8 + i] = (__bf16)p1[i];
    }
    return r;
}

// ---------------------------------------------------------------------------
// Phase 1: C[M,384] = X[M,768] * W^T + bias, bf16 WMMA, WG tile 128x64,
// async double-buffered staging of fp32 tiles.
// ---------------------------------------------------------------------------
#define MT 128
#define NT 64
#define KT 32

__global__ __launch_bounds__(256) void ner_gemm_xp(
    const float* __restrict__ X,
    const float* __restrict__ Wf, const float* __restrict__ Wb,
    const float* __restrict__ biasF, const float* __restrict__ biasB,
    float* __restrict__ XP)
{
    __shared__ float Af[2][MT][KT];   // 2 x 16KB, fp32 row-major [m][k]
    __shared__ float Bf[2][NT][KT];   // 2 x  8KB, fp32 [col][k]

    const int m0 = blockIdx.y * MT;
    const int n0 = blockIdx.x * NT;          // multiple of 64 -> one direction
    const float* W    = (n0 < 192) ? Wf : Wb;
    const float* bias = (n0 < 192) ? biasF : biasB;
    const int gbase   = (n0 < 192) ? n0 : (n0 - 192);

    const int tid  = threadIdx.x;
    const int wave = tid >> 5;
    const int lane = tid & 31;
    const int mi = (wave & 3) * 32;          // wave subtile 32x32
    const int ni = (wave >> 2) * 32;

    v8f acc[2][2] = {};

    const int arow  = lane & 15;
    const int khalf = (lane >> 4) * 8;       // A frag K sub-offset
    const int ksel  = (lane >> 4) * 16;      // unused; B frag also uses khalf

    auto stage = [&](int buf, int k0) {
        #pragma unroll
        for (int it = 0; it < 4; ++it) {     // A: 128x32 fp32 = 1024 x 16B
            int c = tid + it * 256;
            int r = c >> 3, c4 = (c & 7) * 4;
            async_ld16(&Af[buf][r][c4], &X[(size_t)(m0 + r) * Ec + k0 + c4]);
        }
        #pragma unroll
        for (int it = 0; it < 2; ++it) {     // B: 64x32 fp32 = 512 x 16B
            int c = tid + it * 256;
            int r = c >> 3, c4 = (c & 7) * 4;
            async_ld16(&Bf[buf][r][c4], &W[(size_t)(gbase + r) * Ec + k0 + c4]);
        }
    };

    stage(0, 0);
    wait_async0();
    __syncthreads();

    int buf = 0;
    for (int k0 = 0; k0 < Ec; k0 += KT, buf ^= 1) {
        if (k0 + KT < Ec) stage(buf ^ 1, k0 + KT);
        if (k0 + 2 * KT < Ec)                // keep a speculative prefetch alive
            __builtin_prefetch(&X[(size_t)(m0 + (tid & 127)) * Ec + k0 + 2 * KT]);

        v16bf a[2], b[2];
        #pragma unroll
        for (int i = 0; i < 2; ++i) {
            const float* pr = &Af[buf][mi + i * 16 + arow][0];
            a[i] = cvt16(pr + khalf, pr + 16 + khalf);
        }
        #pragma unroll
        for (int j = 0; j < 2; ++j) {
            const float* pc = &Bf[buf][ni + j * 16 + (lane & 15)][khalf];
            b[j] = cvt16(pc, pc + 16);
        }
        #pragma unroll
        for (int i = 0; i < 2; ++i)
            #pragma unroll
            for (int j = 0; j < 2; ++j)
                acc[i][j] = __builtin_amdgcn_wmma_f32_16x16x32_bf16(
                    false, a[i], false, b[j], (short)0, acc[i][j], false, false);

        wait_async0();      // next tile resident before anyone re-fills 'buf'
        __syncthreads();
    }

    // epilogue: + bias, store fp32
    #pragma unroll
    for (int i = 0; i < 2; ++i) {
        #pragma unroll
        for (int j = 0; j < 2; ++j) {
            int col   = ni + j * 16 + (lane & 15);
            int rbase = mi + i * 16 + ((lane >> 4) ? 8 : 0);
            float bv  = bias[gbase + col];
            #pragma unroll
            for (int r = 0; r < 8; ++r)
                XP[(size_t)(m0 + rbase + r) * Gc + n0 + col] = acc[i][j][r] + bv;
        }
    }
}

// ---------------------------------------------------------------------------
// Phase 2: GRU scan. grid (8 batch-slices, 2 dirs), 128 threads (4 waves).
// Per step: gh[16,192] = h_bf16[16,64] @ Whh^T via WMMA, then gates.
// Whh lives in register B-fragments; xp slab for step t+1 is async-prefetched
// into LDS while step t computes.
// ---------------------------------------------------------------------------
__global__ __launch_bounds__(128) void ner_gru_scan(
    const float* __restrict__ xp,        // [M][384], bias already added
    const float* __restrict__ Whh_f, const float* __restrict__ Whh_b,
    const float* __restrict__ bhh_f, const float* __restrict__ bhh_b,
    float* __restrict__ h_all)           // [M][128]
{
    const int slice = blockIdx.x;        // batch rows b0..b0+15
    const int dir   = blockIdx.y;        // 0 fwd, 1 bwd
    const float* Whh = dir ? Whh_b : Whh_f;
    const float* bhh = dir ? bhh_b : bhh_f;

    __shared__ __bf16 hbf[16][64];       //  2KB
    __shared__ float  hf[16][64];        //  4KB
    __shared__ float  gh[16][192];       // 12KB
    __shared__ float  xps[2][16][192];   // 24KB double-buffered xp slab
    __shared__ float  biasH[192];

    const int tid  = threadIdx.x;
    const int wave = tid >> 5;
    const int lane = tid & 31;
    const int b0   = slice * 16;

    for (int i = tid; i < 192; i += 128) biasH[i] = bhh[i];
    for (int i = tid; i < 16 * 64; i += 128) {
        hf[i >> 6][i & 63]  = 0.0f;
        hbf[i >> 6][i & 63] = (__bf16)0.0f;
    }

    // preload Whh as register B fragments straight from global (once)
    const int ksel = (lane >> 4) * 16;
    v16bf Bfrag[3][2];
    #pragma unroll
    for (int f = 0; f < 3; ++f) {
        int g = (wave * 3 + f) * 16 + (lane & 15);
        #pragma unroll
        for (int kk = 0; kk < 2; ++kk) {
            const float* p = &Whh[(size_t)g * Hc + kk * 32 + khalf];
            Bfrag[f][kk] = cvt16(p, p + 16);
        }
    }

    auto stage_xp = [&](int buf, int s) {   // 16 rows x 192 floats = 768 x 16B
        #pragma unroll
        for (int it = 0; it < 6; ++it) {
            int c   = tid + it * 128;
            int row = c / 48;
            int f4  = (c % 48) * 4;
            async_ld16(&xps[buf][row][f4],
                       &xp[((size_t)(b0 + row) * Sc + s) * Gc + dir * 192 + f4]);
        }
    };

    stage_xp(0, dir ? (Sc - 1) : 0);
    wait_async0();
    __syncthreads();

    const int arow  = lane & 15;
    const int khalf = (lane >> 4) * 8;
    const int rb    = (lane >> 4) ? 8 : 0;

    for (int t = 0; t < Sc; ++t) {
        const int s = dir ? (Sc - 1 - t) : t;
        if (t + 1 < Sc) stage_xp((t + 1) & 1, dir ? (Sc - 2 - t) : (t + 1));

        // A fragments from current hidden state
        v16bf a[2];
        #pragma unroll
        for (int kk = 0; kk < 2; ++kk) {
            const float* unused = nullptr; (void)unused;
            // hbf is bf16 in LDS; gather the two 8-half runs directly
            union { v16bf v; uint4 q[2]; } u;
            u.q[0] = *(const uint4*)&hbf[arow][kk * 32 + khalf];
            u.q[1] = *(const uint4*)&hbf[arow][kk * 32 + 16 + khalf];
            a[kk] = u.v;
        }
        #pragma unroll
        for (int f = 0; f < 3; ++f) {
            v8f c = {};
            c = __builtin_amdgcn_wmma_f32_16x16x32_bf16(
                    false, a[0], false, Bfrag[f][0], (short)0, c, false, false);
            c = __builtin_amdgcn_wmma_f32_16x16x32_bf16(
                    false, a[1], false, Bfrag[f][1], (short)0, c, false, false);
            int col = (wave * 3 + f) * 16 + (lane & 15);
            #pragma unroll
            for (int r = 0; r < 8; ++r) gh[rb + r][col] = c[r];
        }
        __syncthreads();   // gh ready

        // gates (torch order r,z,n) reading the LDS-resident xp slab
        for (int i = tid; i < 16 * 64; i += 128) {
            int bi = i >> 6, k = i & 63;
            const float* xpr = &xps[t & 1][bi][0];
            float hr = gh[bi][k]       + biasH[k];
            float hz = gh[bi][64 + k]  + biasH[64 + k];
            float hn = gh[bi][128 + k] + biasH[128 + k];
            float r = 1.0f / (1.0f + __expf(-(xpr[k]      + hr)));
            float z = 1.0f / (1.0f + __expf(-(xpr[64 + k] + hz)));
            float n = tanhf(xpr[128 + k] + r * hn);
            float hnew = (1.0f - z) * n + z * hf[bi][k];
            hf[bi][k]  = hnew;
            hbf[bi][k] = (__bf16)hnew;
            h_all[((size_t)(b0 + bi) * Sc + s) * 128 + dir * Hc + k] = hnew;
        }
        wait_async0();     // t+1 slab resident before its buffer is consumed
        __syncthreads();
    }
}

// ---------------------------------------------------------------------------
// Phase 3: emissions[m][t] = dot(h_all[m], W_dense[t])
// ---------------------------------------------------------------------------
__global__ __launch_bounds__(256) void ner_emissions(
    const float* __restrict__ h_all, const float* __restrict__ Wd,
    float* __restrict__ E)
{
    __shared__ float Wds[Tc][128];
    const int tid = threadIdx.x;
    for (int i = tid; i < Tc * 128; i += 256) Wds[i / 128][i % 128] = Wd[i];
    __syncthreads();

    const int m = blockIdx.x * 16 + (tid >> 4);
    const int t = tid & 15;
    if (t < Tc) {
        const float* h = &h_all[(size_t)m * 128];
        float acc = 0.0f;
        #pragma unroll 8
        for (int k = 0; k < 128; ++k) acc += h[k] * Wds[t][k];
        E[(size_t)m * Tc + t] = acc;
    }
}

// ---------------------------------------------------------------------------
// Phase 4: CRF mean NLL. One thread per batch element, block reduction.
// ---------------------------------------------------------------------------
__global__ __launch_bounds__(128) void ner_crf(
    const float* __restrict__ E, const int* __restrict__ tags,
    const float* __restrict__ trans, const float* __restrict__ st,
    const float* __restrict__ en, float* __restrict__ out)
{
    __shared__ float Ts[Tc][Tc];
    __shared__ float Ss[Tc], En[Tc];
    __shared__ float red[Bc];
    const int tid = threadIdx.x;
    if (tid < Tc * Tc) Ts[tid / Tc][tid % Tc] = trans[tid];
    if (tid < Tc) { Ss[tid] = st[tid]; En[tid] = en[tid]; }
    __syncthreads();

    const int b = tid;
    const float* Eb = &E[(size_t)b * Sc * Tc];
    const int*   tg = &tags[b * Sc];

    int tp = tg[0];
    float num = Ss[tp] + Eb[tp];
    float alpha[Tc];
    #pragma unroll
    for (int t = 0; t < Tc; ++t) alpha[t] = Ss[t] + Eb[t];

    for (int s = 1; s < Sc; ++s) {
        const float* es = &Eb[s * Tc];
        int tc = tg[s];
        num += Ts[tp][tc] + es[tc];
        tp = tc;
        float na[Tc];
        #pragma unroll
        for (int t2 = 0; t2 < Tc; ++t2) {
            float m = -3.0e38f;
            #pragma unroll
            for (int t1 = 0; t1 < Tc; ++t1) m = fmaxf(m, alpha[t1] + Ts[t1][t2]);
            float sum = 0.0f;
            #pragma unroll
            for (int t1 = 0; t1 < Tc; ++t1) sum += __expf(alpha[t1] + Ts[t1][t2] - m);
            na[t2] = m + __logf(sum) + es[t2];
        }
        #pragma unroll
        for (int t = 0; t < Tc; ++t) alpha[t] = na[t];
    }
    num += En[tp];

    float mm = -3.0e38f;
    #pragma unroll
    for (int t = 0; t < Tc; ++t) mm = fmaxf(mm, alpha[t] + En[t]);
    float sum = 0.0f;
    #pragma unroll
    for (int t = 0; t < Tc; ++t) sum += __expf(alpha[t] + En[t] - mm);
    float den = mm + __logf(sum);

    red[tid] = num - den;
    __syncthreads();
    if (tid == 0) {
        float acc = 0.0f;
        for (int i = 0; i < Bc; ++i) acc += red[i];
        out[0] = -(acc / (float)Bc);
    }
}

// ---------------------------------------------------------------------------
extern "C" void kernel_launch(void* const* d_in, const int* in_sizes, int n_in,
                              void* d_out, int out_size, void* d_ws, size_t ws_size,
                              hipStream_t stream)
{
    const float* X     = (const float*)d_in[0];
    const int*   tags  = (const int*)  d_in[1];
    const float* Wih_f = (const float*)d_in[2];
    const float* Whh_f = (const float*)d_in[3];
    const float* bih_f = (const float*)d_in[4];
    const float* bhh_f = (const float*)d_in[5];
    const float* Wih_b = (const float*)d_in[6];
    const float* Whh_b = (const float*)d_in[7];
    const float* bih_b = (const float*)d_in[8];
    const float* bhh_b = (const float*)d_in[9];
    const float* Wd    = (const float*)d_in[10];
    const float* trans = (const float*)d_in[11];
    const float* st    = (const float*)d_in[12];
    const float* en    = (const float*)d_in[13];

    float* ws    = (float*)d_ws;
    float* xp    = ws;                                   // [32768][384]
    float* h_all = xp + (size_t)Mc * Gc;                 // [32768][128]
    float* emis  = h_all + (size_t)Mc * 128;             // [32768][9]

    ner_gemm_xp <<<dim3(Gc / NT, Mc / MT), 256, 0, stream>>>(
        X, Wih_f, Wih_b, bih_f, bih_b, xp);
    ner_gru_scan<<<dim3(Bc / 16, 2), 128, 0, stream>>>(
        xp, Whh_f, Whh_b, bhh_f, bhh_b, h_all);
    ner_emissions<<<Mc / 16, 256, 0, stream>>>(h_all, Wd, emis);
    ner_crf<<<1, Bc, 0, stream>>>(emis, tags, trans, st, en, (float*)d_out);
}
